// TimeAwareRMultiHeadAttention_88450556494367
// MI455X (gfx1250) — compile-verified, matrix operand order checked by reading, not run on hardware
//
#include <hip/hip_runtime.h>
#include <hip/hip_bf16.h>

// ---------------------------------------------------------------------------
// TimeAwareRMultiHeadAttention for gfx1250 (MI455X), wave32 + WMMA f16.
// B=16, N=200, HID=64, NH=2, D=32.  Dominant cost: streaming tk/tv (~328MB),
// loaded with non-temporal hints so the single-use streams bypass-evict L2
// politely while the reused 10MB score/prob working set stays resident.
// ---------------------------------------------------------------------------

#define B_DIM   16
#define N_DIM   200
#define N_PAD   224          // 7 * 32, K padding for ctx WMMA
#define H_DIM   64
#define D_DIM   32
#define NH_DIM  2
#define BH_DIM  (B_DIM * NH_DIM)   // 32
#define NT      13                 // ceil(200/16)

typedef __attribute__((ext_vector_type(16))) _Float16 v16h;
typedef __attribute__((ext_vector_type(8)))  _Float16 h8;
typedef __attribute__((ext_vector_type(8)))  float    v8f;

// Load a 16-half WMMA fragment for one lane: row-major source, per-lane base
// already points at (row, kb). Halves 0..7 <- [kb..kb+7], 8..15 <- [kb+16..kb+23].
__device__ __forceinline__ v16h ld_frag(const _Float16* p) {
  h8 lo = *(const h8*)(p);
  h8 hi = *(const h8*)(p + 16);
  v16h r;
#pragma unroll
  for (int t = 0; t < 8; ++t) { r[t] = lo[t]; r[8 + t] = hi[t]; }
  return r;
}

// ---------------------------------------------------------------------------
// Kernel A: QKV projection (64x64 GEMM per row), head-split f16 staging of
// q and (k+pk), transposed f16 staging of (v+pv), and the 4 rank-1 scalars
// for the order/distance logits.
// grid = B*N blocks of 64 threads (one row each).
// ---------------------------------------------------------------------------
__global__ __launch_bounds__(64) void k_qkv(
    const float* __restrict__ x, const float* __restrict__ pK, const float* __restrict__ pV,
    const float* __restrict__ Wq, const float* __restrict__ bq,
    const float* __restrict__ Wk, const float* __restrict__ bk,
    const float* __restrict__ Wv, const float* __restrict__ bv,
    const float* __restrict__ ow, const float* __restrict__ dw,
    float* __restrict__ qf, _Float16* __restrict__ qh, _Float16* __restrict__ kpkh,
    _Float16* __restrict__ vT,
    float* __restrict__ qo, float* __restrict__ ko,
    float* __restrict__ qd, float* __restrict__ kd) {
  __shared__ float xs[H_DIM];
  __shared__ float r0[H_DIM], r1[H_DIM], r2[H_DIM], r3[H_DIM];
  const int blk = blockIdx.x;               // b*N + i
  const int b = blk / N_DIM, i = blk % N_DIM;
  const int o = threadIdx.x;
  xs[o] = x[(size_t)blk * H_DIM + o];
  __syncthreads();
  float aq = bq[o], ak = bk[o], av = bv[o];
#pragma unroll 8
  for (int c = 0; c < H_DIM; ++c) {
    const float xv = xs[c];
    aq = fmaf(xv, Wq[o * H_DIM + c], aq);
    ak = fmaf(xv, Wk[o * H_DIM + c], ak);
    av = fmaf(xv, Wv[o * H_DIM + c], av);
  }
  const int h = o >> 5, d = o & 31;
  const int bh = b * NH_DIM + h;
  qf[(size_t)blk * H_DIM + o] = aq;
  const size_t hb = ((size_t)bh * N_DIM + i) * D_DIM + d;
  qh[hb]   = (_Float16)aq;
  kpkh[hb] = (_Float16)(ak + pK[(size_t)blk * H_DIM + o]);
  vT[((size_t)bh * D_DIM + d) * N_PAD + i] = (_Float16)(av + pV[(size_t)blk * H_DIM + o]);
  if (i < N_PAD - N_DIM)  // zero the K-padding columns of vT (24 per row)
    vT[((size_t)bh * D_DIM + d) * N_PAD + N_DIM + i] = (_Float16)0.f;
  // rank-1 logit scalars
  r0[o] = aq * ow[d];          // qo: q . order_w[:D]
  r1[o] = ak * ow[D_DIM + d];  // ko: k . order_w[D:]
  r2[o] = aq * dw[d];          // qd
  r3[o] = ak * dw[D_DIM + d];  // kd
  __syncthreads();
  if (o < 8) {
    const int arr = o >> 1, hh = o & 1;
    const float* src = (arr == 0) ? r0 : (arr == 1) ? r1 : (arr == 2) ? r2 : r3;
    float s = 0.f;
    for (int t = 0; t < D_DIM; ++t) s += src[hh * D_DIM + t];
    float* dst = (arr == 0) ? qo : (arr == 1) ? ko : (arr == 2) ? qd : kd;
    dst[(size_t)(b * NH_DIM + hh) * N_DIM + i] = s;
  }
}

// ---------------------------------------------------------------------------
// Kernel B1: base scores via WMMA, S = q . (k+pk), K=32 -> 1 wmma per tile,
// fused order/distance/mask/scale epilogue. 1 wave per 16x16 tile.
// grid = B*NH * 13 * 13 blocks of 32 threads.
// ---------------------------------------------------------------------------
__global__ __launch_bounds__(32) void k_scores(
    const _Float16* __restrict__ qh, const _Float16* __restrict__ kpkh,
    const float* __restrict__ qo, const float* __restrict__ ko,
    const float* __restrict__ qd, const float* __restrict__ kd,
    const float* __restrict__ order_b, const float* __restrict__ dist_b,
    const float* __restrict__ scalar, const float* __restrict__ mask,
    float* __restrict__ scores) {
  const int blk = blockIdx.x;
  const int bh  = blk / (NT * NT);
  const int rem = blk % (NT * NT);
  const int it = rem / NT, jt = rem % NT;
  const int lane = threadIdx.x;
  const int r16 = lane & 15, g16 = lane >> 4;
  const int kb = g16 * 8;
  const int ia = min(it * 16 + r16, N_DIM - 1);   // A row (clamped, no divergence)
  const int ja = min(jt * 16 + r16, N_DIM - 1);   // B col = kpk row
  const v16h a = ld_frag(qh   + ((size_t)bh * N_DIM + ia) * D_DIM + kb);
  const v16h bfrag = ld_frag(kpkh + ((size_t)bh * N_DIM + ja) * D_DIM + kb);
  v8f c = {};
  c = __builtin_amdgcn_wmma_f32_16x16x32_f16(false, a, false, bfrag,
                                             (short)0, c, false, false);
  // fused epilogue
  const int bb = bh >> 1;
  const float ob = order_b[0], db = dist_b[0];
  const float s2 = scalar[0] * scalar[0];
  const float rsD = 0.1767766952966369f;  // 1/sqrt(32)
  const int j = jt * 16 + r16;            // C column (n = lane&15)
  if (j < N_DIM) {
    const float koj = ko[(size_t)bh * N_DIM + j];
    const float kdj = kd[(size_t)bh * N_DIM + j];
#pragma unroll
    for (int r = 0; r < 8; ++r) {
      const int i = it * 16 + g16 * 8 + r;  // C row (m = r + (lane>>4)*8)
      if (i < N_DIM) {
        const float lo = qo[(size_t)bh * N_DIM + i] + koj + ob;
        const float pr = 1.f / (1.f + __expf(-lo));
        const float eo = (j > i) ? __logf(pr + 1e-24f) : __logf(1.f - pr + 1e-24f);
        const float pd = qd[(size_t)bh * N_DIM + i] + kdj + db;
        const float gd = __logf(fabsf((float)(j - i)) + 1.f);
        const float ed = -0.5f * (gd - pd) * (gd - pd) * s2;
        scores[((size_t)bh * N_DIM + i) * N_DIM + j] =
            (c[r] + eo + ed) * rsD + mask[((size_t)bb * N_DIM + i) * N_DIM + j];
      }
    }
  }
}

// ---------------------------------------------------------------------------
// Kernel B2: scores += einsum('bhijd,bhid->bhij', tk, q) / sqrt(D).
// Pure-bandwidth GEMV over the 164MB tk tensor, non-temporal (single use),
// fully coalesced 256B/wave-step.
// grid = B*N blocks of 256 threads; each wave owns j = wid, wid+8, ...
// ---------------------------------------------------------------------------
__global__ __launch_bounds__(256) void k_tk(
    const float* __restrict__ qf, const float* __restrict__ tk,
    float* __restrict__ scores) {
  __shared__ float qs[H_DIM];
  const int blk = blockIdx.x;          // b*N + i
  const int b = blk / N_DIM, i = blk % N_DIM;
  const int t = threadIdx.x;
  if (t < H_DIM) qs[t] = qf[(size_t)blk * H_DIM + t];
  __syncthreads();
  const int wid = t >> 5, lane = t & 31;
  const float rsD = 0.1767766952966369f;
  const float* base = tk + (size_t)blk * N_DIM * H_DIM;
  for (int j = wid; j < N_DIM; j += 8) {
    const float t0 = __builtin_nontemporal_load(base + (size_t)j * H_DIM + lane);
    const float t1 = __builtin_nontemporal_load(base + (size_t)j * H_DIM + lane + 32);
    float p0 = t0 * qs[lane];       // head 0
    float p1 = t1 * qs[lane + 32];  // head 1
    for (int off = 16; off; off >>= 1) {
      p0 += __shfl_xor(p0, off);
      p1 += __shfl_xor(p1, off);
    }
    if (lane == 0) {
      const size_t s0 = ((size_t)(b * NH_DIM + 0) * N_DIM + i) * N_DIM + j;
      const size_t s1 = ((size_t)(b * NH_DIM + 1) * N_DIM + i) * N_DIM + j;
      scores[s0] += p0 * rsD;   // unique (b,h,i,j) per wave-iter: plain RMW
      scores[s1] += p1 * rsD;
    }
  }
}

// ---------------------------------------------------------------------------
// Kernel C: row softmax (wave per row), in-place f32 probs + zero-padded f16
// copy for the WMMA context GEMM. grid = 6400/8 blocks of 256.
// ---------------------------------------------------------------------------
__global__ __launch_bounds__(256) void k_softmax(
    float* __restrict__ scores, _Float16* __restrict__ ph) {
  const int wid = threadIdx.x >> 5, lane = threadIdx.x & 31;
  const int row = blockIdx.x * 8 + wid;       // [0, BH*N)
  if (row >= BH_DIM * N_DIM) return;
  float v[7];
  float m = -1e30f;
#pragma unroll
  for (int p = 0; p < 7; ++p) {
    const int idx = lane + 32 * p;
    v[p] = (idx < N_DIM) ? scores[(size_t)row * N_DIM + idx] : -1e30f;
    m = fmaxf(m, v[p]);
  }
  for (int off = 16; off; off >>= 1) m = fmaxf(m, __shfl_xor(m, off));
  float s = 0.f;
#pragma unroll
  for (int p = 0; p < 7; ++p) { v[p] = __expf(v[p] - m); s += v[p]; }
  for (int off = 16; off; off >>= 1) s += __shfl_xor(s, off);
  const float inv = 1.f / s;
#pragma unroll
  for (int p = 0; p < 7; ++p) {
    const int idx = lane + 32 * p;
    float pv = v[p] * inv;
    if (idx >= N_DIM) pv = 0.f;
    ph[(size_t)row * N_PAD + idx] = (_Float16)pv;
    if (idx < N_DIM) scores[(size_t)row * N_DIM + idx] = pv;
  }
}

// ---------------------------------------------------------------------------
// Kernel D1: ctx = probs @ (v+pv) via WMMA; K tiled 7x32 over padded N.
// grid = B*NH * 13 * 2 blocks of 32 threads (1 wave / 16x16 tile).
// ---------------------------------------------------------------------------
__global__ __launch_bounds__(32) void k_ctx(
    const _Float16* __restrict__ ph, const _Float16* __restrict__ vT,
    float* __restrict__ ctx) {
  const int blk = blockIdx.x;
  const int bh  = blk / (NT * 2);
  const int rem = blk % (NT * 2);
  const int it = rem >> 1, dt = rem & 1;
  const int lane = threadIdx.x;
  const int r16 = lane & 15, g16 = lane >> 4;
  const int kb = g16 * 8;
  const int ia = min(it * 16 + r16, N_DIM - 1);
  v8f c = {};
#pragma unroll
  for (int kc = 0; kc < 7; ++kc) {
    const v16h a = ld_frag(ph + ((size_t)bh * N_DIM + ia) * N_PAD + kc * 32 + kb);
    const v16h bfrag =
        ld_frag(vT + ((size_t)bh * D_DIM + dt * 16 + r16) * N_PAD + kc * 32 + kb);
    c = __builtin_amdgcn_wmma_f32_16x16x32_f16(false, a, false, bfrag,
                                               (short)0, c, false, false);
  }
  const int bb = bh >> 1, h = bh & 1;
  const int d = dt * 16 + r16;
#pragma unroll
  for (int r = 0; r < 8; ++r) {
    const int i = it * 16 + g16 * 8 + r;
    if (i < N_DIM)
      ctx[((size_t)bb * N_DIM + i) * H_DIM + h * D_DIM + d] = c[r];
  }
}

// ---------------------------------------------------------------------------
// Kernel D2: ctx += einsum('bhij,bhijd->bhid', probs, tv).
// Streams the 164MB tv tensor once (non-temporal), coalesced 256B per j-step.
// 4 independent accumulators break the fmac dependency chain (200 = 4*50).
// grid = B*N blocks of 64 threads.
// ---------------------------------------------------------------------------
__global__ __launch_bounds__(64) void k_tv(
    const float* __restrict__ probs, const float* __restrict__ tv,
    float* __restrict__ ctx) {
  const int blk = blockIdx.x;          // b*N + i
  const int b = blk / N_DIM, i = blk % N_DIM;
  const int t = threadIdx.x;
  const int h = t >> 5;
  const float* pr   = probs + ((size_t)(b * NH_DIM + h) * N_DIM + i) * N_DIM;
  const float* base = tv + (size_t)blk * N_DIM * H_DIM + t;
  float a0 = 0.f, a1 = 0.f, a2 = 0.f, a3 = 0.f;
#pragma unroll 2
  for (int j = 0; j < N_DIM; j += 4) {
    const float w0 = pr[j + 0], w1 = pr[j + 1], w2 = pr[j + 2], w3 = pr[j + 3];
    const float e0 = __builtin_nontemporal_load(base + (size_t)(j + 0) * H_DIM);
    const float e1 = __builtin_nontemporal_load(base + (size_t)(j + 1) * H_DIM);
    const float e2 = __builtin_nontemporal_load(base + (size_t)(j + 2) * H_DIM);
    const float e3 = __builtin_nontemporal_load(base + (size_t)(j + 3) * H_DIM);
    a0 = fmaf(w0, e0, a0);
    a1 = fmaf(w1, e1, a1);
    a2 = fmaf(w2, e2, a2);
    a3 = fmaf(w3, e3, a3);
  }
  ctx[(size_t)blk * H_DIM + t] += (a0 + a1) + (a2 + a3);
}

// ---------------------------------------------------------------------------
// Kernel E: h = ctx @ Wd.T + bd + x; LayerNorm(64). grid = B*N x 64.
// ---------------------------------------------------------------------------
__global__ __launch_bounds__(64) void k_out(
    const float* __restrict__ ctx, const float* __restrict__ x,
    const float* __restrict__ Wd, const float* __restrict__ bd,
    const float* __restrict__ g, const float* __restrict__ be,
    float* __restrict__ out) {
  __shared__ float cs[H_DIM];
  __shared__ float red[H_DIM];
  const int blk = blockIdx.x;
  const int o = threadIdx.x;
  cs[o] = ctx[(size_t)blk * H_DIM + o];
  __syncthreads();
  float y = bd[o] + x[(size_t)blk * H_DIM + o];
#pragma unroll 8
  for (int c = 0; c < H_DIM; ++c) y = fmaf(cs[c], Wd[o * H_DIM + c], y);
  red[o] = y; __syncthreads();
  for (int s = 32; s > 0; s >>= 1) { if (o < s) red[o] += red[o + s]; __syncthreads(); }
  const float mu = red[0] * (1.f / H_DIM); __syncthreads();
  const float dv = y - mu;
  red[o] = dv * dv; __syncthreads();
  for (int s = 32; s > 0; s >>= 1) { if (o < s) red[o] += red[o + s]; __syncthreads(); }
  const float var = red[0] * (1.f / H_DIM);
  out[(size_t)blk * H_DIM + o] = dv * rsqrtf(var + 1e-12f) * g[o] + be[o];
}

// ---------------------------------------------------------------------------
// Host launcher: workspace layout (bytes, all 256B aligned):
//   qf32      @        0   (819,200)
//   q_h       @  819,200   (409,600)
//   kpk_h     @1,228,800   (409,600)
//   vT_h      @1,638,400   (458,752)
//   qo/ko/qd/kd @2,097,152 (4 x 25,600)
//   scores    @2,199,552   (10,240,000)  (becomes probs f32 in place)
//   probs_h   @12,439,552  (2,867,200)
//   ctx       @15,306,752  (819,200)     total ~16.1 MB
// ---------------------------------------------------------------------------
extern "C" void kernel_launch(void* const* d_in, const int* in_sizes, int n_in,
                              void* d_out, int out_size, void* d_ws, size_t ws_size,
                              hipStream_t stream) {
  (void)in_sizes; (void)n_in; (void)out_size; (void)ws_size;
  const float* x     = (const float*)d_in[0];
  const float* mask  = (const float*)d_in[1];
  const float* pK    = (const float*)d_in[2];
  const float* pV    = (const float*)d_in[3];
  const float* tk    = (const float*)d_in[4];
  const float* tv    = (const float*)d_in[5];
  const float* Wq    = (const float*)d_in[6];
  const float* bq    = (const float*)d_in[7];
  const float* Wk    = (const float*)d_in[8];
  const float* bk    = (const float*)d_in[9];
  const float* Wv    = (const float*)d_in[10];
  const float* bv    = (const float*)d_in[11];
  const float* Wd    = (const float*)d_in[12];
  const float* bd    = (const float*)d_in[13];
  const float* ow    = (const float*)d_in[14];
  const float* ob    = (const float*)d_in[15];
  const float* dw    = (const float*)d_in[16];
  const float* db    = (const float*)d_in[17];
  const float* sc    = (const float*)d_in[18];
  const float* lng   = (const float*)d_in[19];
  const float* lnb   = (const float*)d_in[20];
  float* out = (float*)d_out;

  char* ws = (char*)d_ws;
  float*    qf     = (float*)   (ws + 0);
  _Float16* qh     = (_Float16*)(ws + 819200);
  _Float16* kpkh   = (_Float16*)(ws + 1228800);
  _Float16* vT     = (_Float16*)(ws + 1638400);
  float*    qo     = (float*)   (ws + 2097152);
  float*    ko     = (float*)   (ws + 2122752);
  float*    qd     = (float*)   (ws + 2148352);
  float*    kd     = (float*)   (ws + 2173952);
  float*    scores = (float*)   (ws + 2199552);
  _Float16* ph     = (_Float16*)(ws + 12439552);
  float*    ctx    = (float*)   (ws + 15306752);

  k_qkv<<<B_DIM * N_DIM, 64, 0, stream>>>(x, pK, pV, Wq, bq, Wk, bk, Wv, bv,
                                          ow, dw, qf, qh, kpkh, vT, qo, ko, qd, kd);
  k_scores<<<BH_DIM * NT * NT, 32, 0, stream>>>(qh, kpkh, qo, ko, qd, kd,
                                                ob, db, sc, mask, scores);
  k_tk<<<B_DIM * N_DIM, 256, 0, stream>>>(qf, tk, scores);
  k_softmax<<<(BH_DIM * N_DIM) / 8, 256, 0, stream>>>(scores, ph);
  k_ctx<<<BH_DIM * NT * 2, 32, 0, stream>>>(ph, vT, ctx);
  k_tv<<<B_DIM * N_DIM, 64, 0, stream>>>(scores, tv, ctx);
  k_out<<<B_DIM * N_DIM, 64, 0, stream>>>(ctx, x, Wd, bd, lng, lnb, out);
}
